// Multi_query_Attention_28372553957595
// MI455X (gfx1250) — matrix operand order repaired and checked by reading, not compiled, not run on hardware
//
#include <hip/hip_runtime.h>
#include <stddef.h>

// ---------------------------------------------------------------------------
// Problem constants (match reference)
// ---------------------------------------------------------------------------
constexpr int EMB = 2048;
constexpr int NH  = 16;
constexpr int HD  = 128;          // head dim
constexpr int BSZ = 4;
constexpr int TSEQ = 2048;
constexpr int MROWS = BSZ * TSEQ; // 8192

typedef __attribute__((ext_vector_type(16))) __bf16 v16bf;
typedef __attribute__((ext_vector_type(8)))  __bf16 v8bf;
typedef __attribute__((ext_vector_type(8)))  float  v8f;
typedef __attribute__((ext_vector_type(4)))  unsigned int u32x4;
typedef __attribute__((ext_vector_type(8)))  int i32x8;
typedef __attribute__((ext_vector_type(4)))  int i32x4;

__device__ inline unsigned short f32_to_bf16(float f) {
  unsigned int u = __float_as_uint(f);
  unsigned int r = 0x7FFFu + ((u >> 16) & 1u);   // round-to-nearest-even
  u += r;
  return (unsigned short)(u >> 16);
}

__device__ inline v8f v8f_zero() {
  v8f z;
#pragma unroll
  for (int i = 0; i < 8; ++i) z[i] = 0.0f;
  return z;
}

// ---------------------------------------------------------------------------
// Tensor Data Mover: 2D tile load (global -> LDS), bf16 elements.
// D# packing per CDNA5 ISA:
//   group0: [1:0]=count=1, [63:32]=lds_addr, [120:64]=global_addr, [127:126]=type=2
//   group1: [17:16]=data_size=1(2B), [79:48]=tensor_dim0, [111:80]=tensor_dim1,
//           [127:112]=tile_dim0, [143:128]=tile_dim1, [207:160]=tensor_dim0_stride
// ---------------------------------------------------------------------------
__device__ inline void tdm_load_2d_bf16(unsigned lds_byte, unsigned long long gaddr,
                                        unsigned tensor_d0, unsigned tensor_d1,
                                        unsigned long long stride0_elems,
                                        unsigned tile_d0, unsigned tile_d1) {
  u32x4 g0;
  g0[0] = 1u;                                              // count=1
  g0[1] = lds_byte;                                        // lds_addr
  g0[2] = (unsigned)(gaddr & 0xFFFFFFFFu);                 // global_addr[31:0]
  g0[3] = (unsigned)((gaddr >> 32) & 0x1FFFFFFu) | (2u << 30); // addr[56:32] | type=2
  i32x8 g1;
  g1[0] = (int)(1u << 16);                                 // data_size=1 -> 2 bytes
  g1[1] = (int)((tensor_d0 & 0xFFFFu) << 16);              // abar=0 | tensor_dim0 lo
  g1[2] = (int)((tensor_d0 >> 16) | ((tensor_d1 & 0xFFFFu) << 16));
  g1[3] = (int)((tensor_d1 >> 16) | ((tile_d0 & 0xFFFFu) << 16));
  g1[4] = (int)(tile_d1 & 0xFFFFu);                        // tile_dim1 | tile_dim2=0
  g1[5] = (int)(stride0_elems & 0xFFFFFFFFu);
  g1[6] = (int)((stride0_elems >> 32) & 0xFFFFu);          // stride0 hi | stride1=0
  g1[7] = 0;
  i32x4 z4 = {0, 0, 0, 0};
#if defined(__clang_major__) && (__clang_major__ >= 23)
  i32x8 z8 = {0, 0, 0, 0, 0, 0, 0, 0};
  __builtin_amdgcn_tensor_load_to_lds(g0, g1, z4, z4, z8, 0);
#else
  __builtin_amdgcn_tensor_load_to_lds(g0, g1, z4, z4, 0);
#endif
}

// ---------------------------------------------------------------------------
// Elementwise converters
// ---------------------------------------------------------------------------
__global__ __launch_bounds__(256) void cvt_f32_bf16(const float* __restrict__ in,
                                                    unsigned short* __restrict__ out,
                                                    int n) {
  int i = blockIdx.x * blockDim.x + threadIdx.x;
  int stride = gridDim.x * blockDim.x;
  for (; i < n; i += stride) out[i] = f32_to_bf16(in[i]);
}

// in [K,N] f32 -> out [N,K] bf16  (weights stored [in,out]; we want W^T rows K-contig)
__global__ __launch_bounds__(256) void cvt_transpose_f32_bf16(const float* __restrict__ in,
                                                              unsigned short* __restrict__ out,
                                                              int K, int N) {
  int i = blockIdx.x * blockDim.x + threadIdx.x;
  int stride = gridDim.x * blockDim.x;
  int total = K * N;
  for (; i < total; i += stride) {
    int kk = i / N, nn = i - kk * N;
    out[(size_t)nn * K + kk] = f32_to_bf16(in[i]);
  }
}

// v bf16 [B*T, D] -> vT bf16 [B, D, T]
__global__ __launch_bounds__(256) void transpose_v_bf16(const unsigned short* __restrict__ in,
                                                        unsigned short* __restrict__ out) {
  int i = blockIdx.x * blockDim.x + threadIdx.x;
  int stride = gridDim.x * blockDim.x;
  int total = MROWS * HD;
  for (; i < total; i += stride) {
    int d = i & (HD - 1);
    int row = i >> 7;                 // b*T + t
    int t = row & (TSEQ - 1);
    int b = row >> 11;
    out[((size_t)b * HD + d) * TSEQ + t] = in[i];
  }
}

// ---------------------------------------------------------------------------
// bf16 WMMA GEMM with TDM-staged weights:
//   C[M,N] = A[M,K] @ Bt[N,K]^T (+bias)
// Block = 8 waves -> 128 rows x 64 cols. Per 32-k-step, wave 0 issues a TDM
// tensor_load_to_lds of the 64x32 bf16 weight slab (double buffered); all
// waves consume B-fragments from LDS, A-fragments direct from global.
// Fragment roles (wave32):
//   A per lane (m=lane&15, g=lane>>4): a[i]=A[m][k0+16g+i]
//   B per lane (n=lane&15, g):         b[i]=Bt[n][k0+16g+i]  (K-contiguous 32B)
//   C/D: lane col n=lane&15, VGPR r -> row r+8g
// ---------------------------------------------------------------------------
template <bool OUT_F32>
__global__ __launch_bounds__(256) void gemm_bf16_wmma(const unsigned short* __restrict__ A,
                                                      const unsigned short* __restrict__ Bt,
                                                      void* __restrict__ Cout,
                                                      const float* __restrict__ bias,
                                                      int M, int N, int K) {
  __shared__ __align__(128) unsigned short btile[2][64 * 32];  // 2 x 4KB slabs

  const int colBlocks = N >> 6;
  const int rb = blockIdx.x / colBlocks;
  const int cb = blockIdx.x - rb * colBlocks;
  const int row0 = rb << 7;        // 128-row block
  const int col0 = cb << 6;        // 64-col block

  const int w = threadIdx.x >> 5;
  const int lane = threadIdx.x & 31;
  const int m = lane & 15, g = lane >> 4;
  const int wrow = row0 + w * 16;

  v8f acc[4];
#pragma unroll
  for (int i = 0; i < 4; ++i) acc[i] = v8f_zero();

  const unsigned short* arow = A + (size_t)(wrow + m) * K;
  const int steps = K >> 5;

  const unsigned long long bt_base =
      (unsigned long long)(size_t)(Bt + (size_t)col0 * K);
  const unsigned lds0 = (unsigned)(size_t)&btile[0][0];
  const unsigned lds1 = (unsigned)(size_t)&btile[1][0];

  // Prologue: stage first weight slab
  if (w == 0) {
    tdm_load_2d_bf16(lds0, bt_base, (unsigned)K, 64u,
                     (unsigned long long)K, 32u, 64u);
  }

  for (int i = 0; i < steps; ++i) {
    const int k0 = i << 5;
    // Issue next slab into the other buffer (overwrites buffer last read 2
    // barriers ago), then wait for the oldest TDM (current buffer) to land.
    if (w == 0 && i + 1 < steps) {
      tdm_load_2d_bf16(((i + 1) & 1) ? lds1 : lds0,
                       bt_base + (unsigned long long)(k0 + 32) * 2u,
                       (unsigned)K, 64u, (unsigned long long)K, 32u, 64u);
    }
    if (i + 1 < steps) __builtin_amdgcn_s_wait_tensorcnt(1);
    else               __builtin_amdgcn_s_wait_tensorcnt(0);
    __syncthreads();

    // A fragment (per-wave private rows): two 16B global loads
    const unsigned short* ap = arow + k0 + 16 * g;
    if (k0 + 32 < K) __builtin_prefetch(ap + 32, 0, 1);
    v8bf alo = *(const v8bf*)ap;
    v8bf ahi = *(const v8bf*)(ap + 8);
    v16bf afrag;
#pragma unroll
    for (int j = 0; j < 8; ++j) { afrag[j] = alo[j]; afrag[8 + j] = ahi[j]; }

    const unsigned short* bl = &btile[i & 1][0];
#pragma unroll
    for (int nt = 0; nt < 4; ++nt) {
      v16bf bfrag = *(const v16bf*)(bl + (nt * 16 + m) * 32 + 16 * g);
      acc[nt] = __builtin_amdgcn_wmma_f32_16x16x32_bf16(
          false, afrag, false, bfrag, (short)0, acc[nt], false, false);
    }
    __syncthreads();   // all waves done reading btile[i&1] before it's reused
  }

#pragma unroll
  for (int nt = 0; nt < 4; ++nt) {
    int col = col0 + nt * 16 + m;
#pragma unroll
    for (int r = 0; r < 8; ++r) {
      int row = wrow + r + 8 * g;
      float v = acc[nt][r];
      if (OUT_F32) {
        ((float*)Cout)[(size_t)row * N + col] = v + (bias ? bias[col] : 0.0f);
      } else {
        ((unsigned short*)Cout)[(size_t)row * N + col] = f32_to_bf16(v);
      }
    }
  }
}

// ---------------------------------------------------------------------------
// Flash MQA attention: one wave handles (b, h, 16-query tile), key tiles of 32.
// Per-wave causal trip counts -> no block barriers; LDS use is intra-wave only.
//   q    : [B*T, EMB] bf16 (col = h*128 + d)
//   kmat : [B*T, HD]  bf16
//   vT   : [B, HD, T] bf16
//   out  : [B*T, EMB] bf16
// ---------------------------------------------------------------------------
__global__ __launch_bounds__(256) void mqa_flash_wmma(const unsigned short* __restrict__ q,
                                                      const unsigned short* __restrict__ kmat,
                                                      const unsigned short* __restrict__ vT,
                                                      unsigned short* __restrict__ attn_out) {
  __shared__ __align__(64) unsigned short p_lds[8][16][32];  // per-wave P staging

  const int w = threadIdx.x >> 5;
  const int wave = blockIdx.x * 8 + w;
  const int QT = TSEQ / 16;                     // 128 query tiles
  if (wave >= BSZ * NH * QT) return;

  const int qt = wave % QT;
  const int h  = (wave / QT) % NH;
  const int b  = wave / (QT * NH);
  const int q0 = qt * 16;

  const int lane = threadIdx.x & 31;
  const int m = lane & 15, g = lane >> 4;       // m: A-row / B-col / C-col role

  // Resident Q fragments: 16 queries x 128 d, split into 4 (16x32) A-fragments
  v16bf qfrag[4];
#pragma unroll
  for (int ks = 0; ks < 4; ++ks) {
    const unsigned short* qp =
        q + (size_t)(b * TSEQ + q0 + m) * EMB + h * HD + ks * 32 + 16 * g;
    v8bf lo = *(const v8bf*)qp;
    v8bf hi = *(const v8bf*)(qp + 8);
#pragma unroll
    for (int i = 0; i < 8; ++i) { qfrag[ks][i] = lo[i]; qfrag[ks][8 + i] = hi[i]; }
  }

  float mrow[8], lrow[8];
  v8f o[8];
#pragma unroll
  for (int r = 0; r < 8; ++r) { mrow[r] = -1e30f; lrow[r] = 0.0f; }
#pragma unroll
  for (int dt = 0; dt < 8; ++dt) o[dt] = v8f_zero();

  const float scale = 0.08838834764831845f;     // 1/sqrt(128)
  const int kend = q0 + 15;

  for (int kb = 0; kb <= kend; kb += 32) {
    // ---- S = Q @ K^T for 16 queries x 32 keys (two 16x16 n-tiles) ----
    v8f s[2];
    s[0] = v8f_zero(); s[1] = v8f_zero();
#pragma unroll
    for (int t = 0; t < 2; ++t) {
      const int key = kb + t * 16 + m;          // B-fragment column role
#pragma unroll
      for (int ks = 0; ks < 4; ++ks) {
        const unsigned short* kp =
            kmat + (size_t)(b * TSEQ + key) * HD + ks * 32 + 16 * g;
        v16bf kfrag = *(const v16bf*)kp;
        s[t] = __builtin_amdgcn_wmma_f32_16x16x32_bf16(
            false, qfrag[ks], false, kfrag, (short)0, s[t], false, false);
      }
    }

    // ---- scale + causal mask (C-layout: row r+8g, col = kb + t*16 + m) ----
#pragma unroll
    for (int t = 0; t < 2; ++t) {
      const int key = kb + t * 16 + m;
#pragma unroll
      for (int r = 0; r < 8; ++r) {
        float val = s[t][r] * scale;
        if (key > q0 + r + 8 * g) val = -1e30f;
        s[t][r] = val;
      }
    }

    // ---- online softmax: half-wave (width-16) row reductions ----
#pragma unroll
    for (int r = 0; r < 8; ++r) {
      float tm = fmaxf(s[0][r], s[1][r]);
#pragma unroll
      for (int off = 8; off >= 1; off >>= 1) tm = fmaxf(tm, __shfl_xor(tm, off, 16));
      float mnew  = fmaxf(mrow[r], tm);
      float alpha = __expf(mrow[r] - mnew);
      mrow[r] = mnew;
      float p0 = __expf(s[0][r] - mnew);
      float p1 = __expf(s[1][r] - mnew);
      s[0][r] = p0; s[1][r] = p1;
      float ts = p0 + p1;
#pragma unroll
      for (int off = 8; off >= 1; off >>= 1) ts += __shfl_xor(ts, off, 16);
      lrow[r] = lrow[r] * alpha + ts;
#pragma unroll
      for (int dt = 0; dt < 8; ++dt) o[dt][r] *= alpha;
    }

    // ---- P: C-layout -> A-layout via LDS (intra-wave; no block barrier) ----
    unsigned short* pw = &p_lds[w][0][0];
#pragma unroll
    for (int r = 0; r < 8; ++r) {
      int row = r + 8 * g;
      pw[row * 32 + m]      = f32_to_bf16(s[0][r]);
      pw[row * 32 + 16 + m] = f32_to_bf16(s[1][r]);
    }
    asm volatile("s_wait_dscnt 0" ::: "memory");
    const unsigned short* pr = pw + m * 32 + 16 * g;
    v8bf plo = *(const v8bf*)pr;
    v8bf phi = *(const v8bf*)(pr + 8);
    v16bf pfrag;
#pragma unroll
    for (int i = 0; i < 8; ++i) { pfrag[i] = plo[i]; pfrag[8 + i] = phi[i]; }

    // ---- O += P @ V  (8 d-tiles of 16; V^T rows give contiguous B-fragments) ----
#pragma unroll
    for (int dt = 0; dt < 8; ++dt) {
      const unsigned short* vp =
          vT + ((size_t)b * HD + dt * 16 + m) * TSEQ + kb + 16 * g;
      v16bf vfrag = *(const v16bf*)vp;
      o[dt] = __builtin_amdgcn_wmma_f32_16x16x32_bf16(
          false, pfrag, false, vfrag, (short)0, o[dt], false, false);
    }
  }

  // ---- epilogue: O / l, write bf16 [B*T, EMB] ----
#pragma unroll
  for (int dt = 0; dt < 8; ++dt) {
#pragma unroll
    for (int r = 0; r < 8; ++r) {
      float val = o[dt][r] / lrow[r];
      int row = b * TSEQ + q0 + r + 8 * g;
      attn_out[(size_t)row * EMB + h * HD + dt * 16 + m] = f32_to_bf16(val);
    }
  }
}

// ---------------------------------------------------------------------------
// Host orchestration
// ---------------------------------------------------------------------------
extern "C" void kernel_launch(void* const* d_in, const int* in_sizes, int n_in,
                              void* d_out, int out_size, void* d_ws, size_t ws_size,
                              hipStream_t stream) {
  (void)in_sizes; (void)n_in; (void)out_size; (void)ws_size;
  const float* x  = (const float*)d_in[0];
  const float* Wq = (const float*)d_in[1];
  const float* Wk = (const float*)d_in[2];
  const float* Wv = (const float*)d_in[3];
  const float* Wo = (const float*)d_in[4];
  const float* bo = (const float*)d_in[5];
  float* out = (float*)d_out;

  char* p = (char*)d_ws;
  auto alloc = [&](size_t bytes) {
    char* r = p;
    p += (bytes + 255) & ~(size_t)255;
    return r;
  };
  unsigned short* xb  = (unsigned short*)alloc((size_t)MROWS * EMB * 2); // 33.5 MB
  unsigned short* wqT = (unsigned short*)alloc((size_t)EMB * EMB * 2);   //  8.4 MB
  unsigned short* wkT = (unsigned short*)alloc((size_t)HD * EMB * 2);
  unsigned short* wvT = (unsigned short*)alloc((size_t)HD * EMB * 2);
  unsigned short* woT = (unsigned short*)alloc((size_t)EMB * EMB * 2);
  unsigned short* qb  = (unsigned short*)alloc((size_t)MROWS * EMB * 2); // 33.5 MB
  unsigned short* kb  = (unsigned short*)alloc((size_t)MROWS * HD * 2);
  unsigned short* vb  = (unsigned short*)alloc((size_t)MROWS * HD * 2);
  unsigned short* vTb = (unsigned short*)alloc((size_t)MROWS * HD * 2);
  unsigned short* ab  = xb; // reuse: x_bf16 dead once projections complete

  // 1) conversions
  cvt_f32_bf16<<<8192, 256, 0, stream>>>(x, xb, MROWS * EMB);
  cvt_transpose_f32_bf16<<<4096, 256, 0, stream>>>(Wq, wqT, EMB, EMB);
  cvt_transpose_f32_bf16<<<512, 256, 0, stream>>>(Wk, wkT, EMB, HD);
  cvt_transpose_f32_bf16<<<512, 256, 0, stream>>>(Wv, wvT, EMB, HD);
  cvt_transpose_f32_bf16<<<4096, 256, 0, stream>>>(Wo, woT, EMB, EMB);

  // 2) projections (bf16 out): q = x@Wq, k = x@Wk, v = x@Wv
  {
    int blocks = (MROWS / 128) * (EMB / 64);   // 2048
    gemm_bf16_wmma<false><<<blocks, 256, 0, stream>>>(xb, wqT, qb, nullptr,
                                                      MROWS, EMB, EMB);
  }
  {
    int blocks = (MROWS / 128) * (HD / 64);    // 128
    gemm_bf16_wmma<false><<<blocks, 256, 0, stream>>>(xb, wkT, kb, nullptr,
                                                      MROWS, HD, EMB);
    gemm_bf16_wmma<false><<<blocks, 256, 0, stream>>>(xb, wvT, vb, nullptr,
                                                      MROWS, HD, EMB);
  }

  // 3) V^T for contiguous PV B-fragments
  transpose_v_bf16<<<1024, 256, 0, stream>>>(vb, vTb);

  // 4) flash attention -> ab (aliases xb)
  {
    int waves = BSZ * NH * (TSEQ / 16);        // 8192
    mqa_flash_wmma<<<waves / 8, 256, 0, stream>>>(qb, kb, vTb, ab);
  }

  // 5) output projection (f32 out + bias)
  {
    int blocks = (MROWS / 128) * (EMB / 64);   // 2048
    gemm_bf16_wmma<true><<<blocks, 256, 0, stream>>>(ab, woT, out, bo,
                                                     MROWS, EMB, EMB);
  }
}
